// FusedKOSLMSSMModule_47528108098140
// MI455X (gfx1250) — compile-verified, matrix operand order checked by reading, not run on hardware
//
#include <hip/hip_runtime.h>
#include <hip/hip_bf16.h>
#include <math.h>

// ---------------------------------------------------------------------------
// FusedKOSLMSSM for MI455X (gfx1250, wave32, WMMA bf16 + async-to-LDS path)
// ---------------------------------------------------------------------------

typedef __bf16 bf16_t;
typedef __attribute__((ext_vector_type(16))) __bf16 v16bf;
typedef __attribute__((ext_vector_type(8)))  __bf16 v8bf;
typedef __attribute__((ext_vector_type(8)))  float  v8f;
typedef __attribute__((ext_vector_type(4)))  int    v4i;

#define K_SCALE  0.9f
#define M_SCALE  1.5f
#define MAX_NORM 100.0f
#define EPSF     1e-12f

#if defined(__gfx1250__) && \
    __has_builtin(__builtin_amdgcn_global_load_async_to_lds_b128) && \
    __has_builtin(__builtin_amdgcn_s_wait_asynccnt)
#define USE_ASYNC_LDS 1
#else
#define USE_ASYNC_LDS 0
#endif

#define AS3 __attribute__((address_space(3)))

// ---------------------------------------------------------------------------
// small prep kernels
// ---------------------------------------------------------------------------
__global__ void cvt_f32_to_bf16(const float* __restrict__ in,
                                bf16_t* __restrict__ out, int n) {
  int i = blockIdx.x * blockDim.x + threadIdx.x;
  int stride = gridDim.x * blockDim.x;
  for (; i < n; i += stride) out[i] = (bf16_t)in[i];
}

__global__ void sigmoid_kernel(const float* __restrict__ in,
                               float* __restrict__ out, int n) {
  int i = blockIdx.x * blockDim.x + threadIdx.x;
  if (i < n) out[i] = 1.0f / (1.0f + __expf(-in[i]));
}

// innov = gates[:, :H] + z_bias - c_prev   (stored bf16 for WMMA A-matrix)
__global__ void innov_kernel(const float* __restrict__ gates,
                             const float* __restrict__ z_bias,
                             const float* __restrict__ c_prev,
                             bf16_t* __restrict__ innov,
                             int B, int H) {
  size_t i = (size_t)blockIdx.x * blockDim.x + threadIdx.x;
  size_t n = (size_t)B * H;
  size_t stride = (size_t)gridDim.x * blockDim.x;
  for (; i < n; i += stride) {
    size_t b = i / H;
    int    h = (int)(i - b * H);
    float v = gates[b * (size_t)(2 * H) + h] + z_bias[h] - c_prev[i];
    innov[i] = (bf16_t)v;
  }
}

// ---------------------------------------------------------------------------
// WMMA bf16 GEMM:  out[M,N] = epilogue( A[M,K] * W[N,K]^T + bias[N] )
//   EPI=0 : exact GELU, store bf16
//   EPI=1 : K_SCALE * sigmoid, store f32
// Block: 256 threads (8 waves). Tile: BM=128, BN=128, BK=32, double-buffered.
// Wave grid 4(M) x 2(N); each wave owns a 32x64 tile = 2x4 WMMA 16x16 tiles.
// Staging: GLOBAL_LOAD_ASYNC_TO_LDS_B128 (ASYNCcnt) overlapped with WMMA.
// ---------------------------------------------------------------------------
template <int EPI>
__global__ __launch_bounds__(256)
void gemm_wmma_bf16(const bf16_t* __restrict__ A,   // [M,K] row-major
                    const bf16_t* __restrict__ W,   // [N,K] row-major (weights)
                    const float*  __restrict__ bias,// [N]
                    void* __restrict__ outv,
                    int M, int N, int K) {
  constexpr int BM = 128, BN = 128, BK = 32, PAD = 8;
  __shared__ bf16_t As[2][BM][BK + PAD];  // 2 x 10240 B
  __shared__ bf16_t Bs[2][BN][BK + PAD];  // 2 x 10240 B

  const int tid   = threadIdx.x;
  const int lane  = tid & 31;
  const int wave  = tid >> 5;
  const int waveM = wave & 3;   // 0..3
  const int waveN = wave >> 2;  // 0..1
  const int l16   = lane & 15;
  const int half  = lane >> 4;  // 0/1 : K-halves per ISA 16-bit A layout

  const int mBase = blockIdx.y * BM;
  const int nBase = blockIdx.x * BN;

  // this thread's two staging chunks (16B each, for A and for B tiles)
  const int c0 = tid;            // chunk ids: tid and tid+256
  const int c1 = tid + 256;
  const int r0 = c0 >> 2, kc0 = (c0 & 3) * 8;
  const int r1 = c1 >> 2, kc1 = (c1 & 3) * 8;

  v8f acc[2][4];
#pragma unroll
  for (int mf = 0; mf < 2; ++mf)
#pragma unroll
    for (int nf = 0; nf < 4; ++nf) acc[mf][nf] = {};

#if USE_ASYNC_LDS
  // --- async DMA staging of one k-slice into LDS buffer `buf`
  auto stage = [&](int buf, int k0) {
    __builtin_amdgcn_global_load_async_to_lds_b128(
        (v4i*)(A + (size_t)(mBase + r0) * K + k0 + kc0),
        (AS3 v4i*)(&As[buf][r0][kc0]), 0, 0);
    __builtin_amdgcn_global_load_async_to_lds_b128(
        (v4i*)(A + (size_t)(mBase + r1) * K + k0 + kc1),
        (AS3 v4i*)(&As[buf][r1][kc1]), 0, 0);
    __builtin_amdgcn_global_load_async_to_lds_b128(
        (v4i*)(W + (size_t)(nBase + r0) * K + k0 + kc0),
        (AS3 v4i*)(&Bs[buf][r0][kc0]), 0, 0);
    __builtin_amdgcn_global_load_async_to_lds_b128(
        (v4i*)(W + (size_t)(nBase + r1) * K + k0 + kc1),
        (AS3 v4i*)(&Bs[buf][r1][kc1]), 0, 0);
  };
#else
  // --- fallback: synchronous VGPR-bounce staging
  auto stage = [&](int buf, int k0) {
    *(v8bf*)(&As[buf][r0][kc0]) =
        *(const v8bf*)(A + (size_t)(mBase + r0) * K + k0 + kc0);
    *(v8bf*)(&As[buf][r1][kc1]) =
        *(const v8bf*)(A + (size_t)(mBase + r1) * K + k0 + kc1);
    *(v8bf*)(&Bs[buf][r0][kc0]) =
        *(const v8bf*)(W + (size_t)(nBase + r0) * K + k0 + kc0);
    *(v8bf*)(&Bs[buf][r1][kc1]) =
        *(const v8bf*)(W + (size_t)(nBase + r1) * K + k0 + kc1);
  };
#endif

  const int nsteps = K / BK;

  // prologue: fill buffer 0
  stage(0, 0);

  for (int i = 0; i < nsteps; ++i) {
    const int cur = i & 1;
#if USE_ASYNC_LDS
    __builtin_amdgcn_s_wait_asynccnt(0);  // my async writes for `cur` landed
#endif
    __syncthreads();                      // everyone's writes visible

    // overlap: kick off next slice's DMA before computing on `cur`
    if (i + 1 < nsteps) stage(cur ^ 1, (i + 1) * BK);

    // --- build fragments per ISA 16-bit layout:
    // lane holds row (M or N) = l16; K chunks at half*8 and 16+half*8
    v16bf aF[2];
#pragma unroll
    for (int mf = 0; mf < 2; ++mf) {
      int r = waveM * 32 + mf * 16 + l16;
      v8bf lo = *(const v8bf*)(&As[cur][r][half * 8]);
      v8bf hi = *(const v8bf*)(&As[cur][r][16 + half * 8]);
#pragma unroll
      for (int j = 0; j < 8; ++j) { aF[mf][j] = lo[j]; aF[mf][8 + j] = hi[j]; }
    }
    v16bf bF[4];
#pragma unroll
    for (int nf = 0; nf < 4; ++nf) {
      int r = waveN * 64 + nf * 16 + l16;
      v8bf lo = *(const v8bf*)(&Bs[cur][r][half * 8]);
      v8bf hi = *(const v8bf*)(&Bs[cur][r][16 + half * 8]);
#pragma unroll
      for (int j = 0; j < 8; ++j) { bF[nf][j] = lo[j]; bF[nf][8 + j] = hi[j]; }
    }

    // --- 8 WMMAs, f32 accumulate
#pragma unroll
    for (int mf = 0; mf < 2; ++mf)
#pragma unroll
      for (int nf = 0; nf < 4; ++nf)
        acc[mf][nf] = __builtin_amdgcn_wmma_f32_16x16x32_bf16(
            false, aF[mf], false, bF[nf], (short)0, acc[mf][nf], false, false);

    __syncthreads();  // all reads of `cur` done before it is re-staged
  }

  // --- epilogue: C/D layout = lane l16 is N col, VGPR v is M row
  //     (row = half*8 + v within the 16x16 tile)
#pragma unroll
  for (int mf = 0; mf < 2; ++mf) {
#pragma unroll
    for (int nf = 0; nf < 4; ++nf) {
      int col = nBase + waveN * 64 + nf * 16 + l16;
      float bv = bias[col];
#pragma unroll
      for (int v = 0; v < 8; ++v) {
        int row = mBase + waveM * 32 + mf * 16 + half * 8 + v;
        float x = acc[mf][nf][v] + bv;
        if (EPI == 0) {
          // exact GELU: 0.5*x*(1+erf(x/sqrt(2))) -> bf16
          float y = 0.5f * x * (1.0f + erff(x * 0.70710678118654752f));
          ((bf16_t*)outv)[(size_t)row * N + col] = (bf16_t)y;
        } else {
          // K = K_SCALE * sigmoid(x) -> f32
          float y = K_SCALE / (1.0f + __expf(-x));
          ((float*)outv)[(size_t)row * N + col] = y;
        }
      }
    }
  }
}

// ---------------------------------------------------------------------------
// Per-row finalize: one block per batch row.
//   innov recomputed in f32, c update, row sum-of-squares, norm clip,
//   out_h / out_c / out_A.
// ---------------------------------------------------------------------------
__global__ __launch_bounds__(256)
void finalize_kernel(const float* __restrict__ gates,
                     const float* __restrict__ c_prev,
                     const float* __restrict__ A_prev,
                     const float* __restrict__ a_sig,   // sigmoid(A_base) [H]
                     const float* __restrict__ Kmat,    // [B,H]
                     const float* __restrict__ z_bias,
                     const float* __restrict__ m_bias,
                     float* __restrict__ out_h,
                     float* __restrict__ out_c,
                     float* __restrict__ out_A,
                     int H) {
  const int b   = blockIdx.x;
  const int tid = threadIdx.x;
  const int PER = H / 256;  // 4 for H=1024
  __shared__ float partial[8];

  float cns[8];
  float Kv[8];
  float ss = 0.0f;

  const size_t rowH  = (size_t)b * H;
  const size_t row2H = (size_t)b * (2 * H);

#pragma unroll 4
  for (int i = 0; i < PER; ++i) {
    int h = tid + i * 256;
    float cp  = c_prev[rowH + h];
    float kv  = Kmat[rowH + h];
    float inn = gates[row2H + h] + z_bias[h] - cp;
    float c   = cp + kv * inn;
    cns[i] = c;
    Kv[i]  = kv;
    ss += c * c;
  }

  // wave32 reduction, then cross-wave via LDS
#pragma unroll
  for (int off = 16; off > 0; off >>= 1) ss += __shfl_xor(ss, off, 32);
  if ((tid & 31) == 0) partial[tid >> 5] = ss;
  __syncthreads();
  if (tid == 0) {
    float t = 0.0f;
#pragma unroll
    for (int w = 0; w < 8; ++w) t += partial[w];
    partial[0] = t;
  }
  __syncthreads();

  const float norm  = sqrtf(partial[0] + EPSF);
  const float scale = fminf(1.0f, MAX_NORM / norm);

#pragma unroll 4
  for (int i = 0; i < PER; ++i) {
    int h = tid + i * 256;
    float oc = cns[i] * scale;
    float m  = M_SCALE * tanhf(gates[row2H + H + h] + m_bias[h]);
    float a  = a_sig[h];
    out_c[rowH + h] = oc;
    out_h[rowH + h] = m * tanhf(oc);
    out_A[rowH + h] = a * A_prev[rowH + h] + (1.0f - a) * Kv[i];
  }
}

// ---------------------------------------------------------------------------
// launch
// ---------------------------------------------------------------------------
extern "C" void kernel_launch(void* const* d_in, const int* in_sizes, int n_in,
                              void* d_out, int out_size, void* d_ws,
                              size_t ws_size, hipStream_t stream) {
  const float* gates  = (const float*)d_in[0];  // [B, 2H]
  const float* c_prev = (const float*)d_in[1];  // [B, H]
  const float* A_prev = (const float*)d_in[2];  // [B, H]
  const float* A_base = (const float*)d_in[3];  // [H]
  const float* W1     = (const float*)d_in[4];  // [3H, H]
  const float* b1     = (const float*)d_in[5];  // [3H]
  const float* W2     = (const float*)d_in[6];  // [H, 3H]
  const float* b2     = (const float*)d_in[7];  // [H]
  const float* z_bias = (const float*)d_in[8];  // [H]
  const float* m_bias = (const float*)d_in[9];  // [H]

  const int H  = in_sizes[3];
  const int B  = in_sizes[1] / H;
  const int H3 = 3 * H;

  float* out   = (float*)d_out;
  float* out_h = out;
  float* out_c = out + (size_t)B * H;
  float* out_A = out + 2 * (size_t)B * H;

  // workspace carve-up (256B aligned)
  char*  ws  = (char*)d_ws;
  size_t off = 0;
  auto carve = [&](size_t bytes) -> void* {
    void* p = ws + off;
    off += (bytes + 255) & ~(size_t)255;
    return p;
  };
  bf16_t* w1_bf    = (bf16_t*)carve((size_t)H3 * H * sizeof(bf16_t));
  bf16_t* w2_bf    = (bf16_t*)carve((size_t)H * H3 * sizeof(bf16_t));
  bf16_t* innov_bf = (bf16_t*)carve((size_t)B * H * sizeof(bf16_t));
  bf16_t* g_bf     = (bf16_t*)carve((size_t)B * H3 * sizeof(bf16_t));
  float*  Kmat     = (float*)carve((size_t)B * H * sizeof(float));
  float*  a_sig    = (float*)carve((size_t)H * sizeof(float));

  // 1) prep: weight conversion + sigmoid(A_base) + innov (bf16)
  {
    int n1 = H3 * H;
    cvt_f32_to_bf16<<<2048, 256, 0, stream>>>(W1, w1_bf, n1);
    cvt_f32_to_bf16<<<2048, 256, 0, stream>>>(W2, w2_bf, n1);
    sigmoid_kernel<<<(H + 255) / 256, 256, 0, stream>>>(A_base, a_sig, H);
    innov_kernel<<<4096, 256, 0, stream>>>(gates, z_bias, c_prev, innov_bf,
                                           B, H);
  }

  // 2) GEMM1 + exact GELU -> g (bf16):  [B,H] x [3H,H]^T = [B,3H]
  {
    dim3 grid(H3 / 128, B / 128);
    gemm_wmma_bf16<0><<<grid, 256, 0, stream>>>(innov_bf, w1_bf, b1,
                                                (void*)g_bf, B, H3, H);
  }

  // 3) GEMM2 + K_SCALE*sigmoid -> Kmat (f32):  [B,3H] x [H,3H]^T = [B,H]
  {
    dim3 grid(H / 128, B / 128);
    gemm_wmma_bf16<1><<<grid, 256, 0, stream>>>(g_bf, w2_bf, b2,
                                                (void*)Kmat, B, H, H3);
  }

  // 4) per-row state update + norm clip + outputs
  finalize_kernel<<<B, 256, 0, stream>>>(gates, c_prev, A_prev, a_sig, Kmat,
                                         z_bias, m_bias, out_h, out_c, out_A,
                                         H);
}
